// ConfigurableAttention_48387101557373
// MI455X (gfx1250) — compile-verified, matrix operand order checked by reading, not run on hardware
//
#include <hip/hip_runtime.h>

typedef __attribute__((ext_vector_type(16))) _Float16 v16h;
typedef __attribute__((ext_vector_type(8)))  _Float16 v8h;
typedef __attribute__((ext_vector_type(8)))  float    v8f;

#define LSEQ 2048
#define DM   512
#define NH   8
#define DH   64
#define NB   4
#define MTOT (NB*LSEQ)   /* 8192 */
#define WIN  64

// Build a 16-half A/B fragment from two contiguous 8-half (16B) chunks.
// CDNA5 16-bit A-matrix 16x32 layout: lane holds K = [base..base+7] in v0..3
// and K = [base+16..base+23] in v4..7, base = (lane>=16)?8:0.
__device__ __forceinline__ v16h make_frag(const _Float16* p0, const _Float16* p1) {
    v8h lo = *(const v8h*)p0;
    v8h hi = *(const v8h*)p1;
    return __builtin_shufflevector(lo, hi, 0,1,2,3,4,5,6,7,8,9,10,11,12,13,14,15);
}
__device__ __forceinline__ v16h load_frag(const _Float16* row, int base) {
    return make_frag(row + base, row + base + 16);
}
__device__ __forceinline__ v8f wmma_f16(v16h a, v16h b, v8f c) {
    return __builtin_amdgcn_wmma_f32_16x16x32_f16(false, a, false, b, (short)0, c, false, false);
}

// Load one k-step's 2 A-fragments + 4 B-fragments (12 b128 loads).
__device__ __forceinline__ void load_ab(v16h a[2], v16h b[4],
                                        const _Float16* arow0, const _Float16* arow1,
                                        const _Float16* const brow[4], int k, int base) {
    a[0] = load_frag(arow0 + k, base);
    a[1] = load_frag(arow1 + k, base);
#pragma unroll
    for (int t = 0; t < 4; ++t) b[t] = load_frag(brow[t] + k, base);
}
// 8 WMMAs accumulating into acc.
__device__ __forceinline__ void mma_step(v8f acc[2][4], const v16h a[2], const v16h b[4]) {
#pragma unroll
    for (int rb = 0; rb < 2; ++rb)
#pragma unroll
        for (int t = 0; t < 4; ++t)
            acc[rb][t] = wmma_f16(a[rb], b[t], acc[rb][t]);
}
// First k-step: C is a fresh zero vector -> lowers to the inline-0 C operand.
__device__ __forceinline__ void mma_first(v8f acc[2][4], const v16h a[2], const v16h b[4]) {
#pragma unroll
    for (int rb = 0; rb < 2; ++rb)
#pragma unroll
        for (int t = 0; t < 4; ++t) {
            v8f z = {};
            acc[rb][t] = wmma_f16(a[rb], b[t], z);
        }
}

// ---------------- conversion kernels ----------------
__global__ void cvt_x_kernel(const float* __restrict__ x, _Float16* __restrict__ xh, int n) {
    int i = blockIdx.x * blockDim.x + threadIdx.x;
    int stride = gridDim.x * blockDim.x;
    for (; i < n; i += stride) xh[i] = (_Float16)x[i];
}

// W[k][n] (fp32, row-major K x N) -> W^T[n][k] (f16). 4*512*512 elements.
__global__ void cvt_w_kernel(const float* __restrict__ Wq, const float* __restrict__ Wk,
                             const float* __restrict__ Wv, const float* __restrict__ Wo,
                             _Float16* __restrict__ wqkvT, _Float16* __restrict__ woT) {
    int tid = blockIdx.x * 256 + threadIdx.x;
    int p   = tid >> 18;
    int rem = tid & ((1 << 18) - 1);
    int nn  = rem >> 9;
    int kk  = rem & 511;
    const float* W = (p == 0) ? Wq : (p == 1) ? Wk : (p == 2) ? Wv : Wo;
    _Float16 v = (_Float16)W[kk * DM + nn];
    if (p < 3) wqkvT[(size_t)p * DM * DM + nn * DM + kk] = v;
    else       woT[nn * DM + kk] = v;
}

// ---------------- fused QKV projection GEMM ----------------
// grid (N/64=8, M/128=64, 3). 128 threads = 4 waves; wave -> 32 rows x 64 cols.
// Ping-pong pipelined: no register copies, first step uses inline-0 C.
__global__ void __launch_bounds__(128) qkv_gemm_kernel(
    const _Float16* __restrict__ xh, const _Float16* __restrict__ wqkvT,
    const float* __restrict__ bq, const float* __restrict__ bk, const float* __restrict__ bv,
    _Float16* __restrict__ qh, _Float16* __restrict__ kh, _Float16* __restrict__ vT) {
    const int lane   = threadIdx.x & 31;
    const int wave   = threadIdx.x >> 5;
    const int lane15 = lane & 15;
    const int halfw  = lane >> 4;
    const int base   = halfw * 8;
    const int n0     = blockIdx.x * 64;
    const int m0     = blockIdx.y * 128 + wave * 32;
    const int p      = blockIdx.z;

    const _Float16* wT    = wqkvT + (size_t)p * DM * DM;
    const _Float16* arow0 = xh + (size_t)(m0 + lane15) * DM;
    const _Float16* arow1 = arow0 + (size_t)16 * DM;
    const _Float16* brow[4];
#pragma unroll
    for (int t = 0; t < 4; ++t)
        brow[t] = wT + (size_t)(n0 + t * 16 + lane15) * DM;

    v8f acc[2][4];
    v16h a0[2], b0[4], a1[2], b1[4];
    load_ab(a0, b0, arow0, arow1, brow, 0, base);   // k = 0
    load_ab(a1, b1, arow0, arow1, brow, 32, base);  // k = 32
    mma_first(acc, a0, b0);                         // k = 0, C = inline 0

    for (int k0 = 64; k0 < DM; k0 += 64) {          // 7 iterations
        load_ab(a0, b0, arow0, arow1, brow, k0, base);       // prefetch k0
        mma_step(acc, a1, b1);                               // compute k0-32
        load_ab(a1, b1, arow0, arow1, brow, k0 + 32, base);  // prefetch k0+32
        mma_step(acc, a0, b0);                               // compute k0
    }
    mma_step(acc, a1, b1);                          // drain k = 480

    const float* bias = (p == 0) ? bq : (p == 1) ? bk : bv;
#pragma unroll
    for (int t = 0; t < 4; ++t) {
        int col = n0 + t * 16 + lane15;
        int h = col >> 6, d = col & 63;
        float bcol = bias[col];
#pragma unroll
        for (int rb = 0; rb < 2; ++rb)
#pragma unroll
            for (int r = 0; r < 8; ++r) {
                int m = m0 + rb * 16 + r + halfw * 8;
                int b2 = m >> 11, l = m & (LSEQ - 1);
                float val = acc[rb][t][r] + bcol;
                size_t bhh = (size_t)(b2 * NH + h);
                if (p == 0)      qh[(bhh * LSEQ + l) * DH + d] = (_Float16)(val * 0.125f); // 1/sqrt(64)
                else if (p == 1) kh[(bhh * LSEQ + l) * DH + d] = (_Float16)val;
                else             vT[(bhh * DH + d) * LSEQ + l] = (_Float16)val;            // V^T [b,h,d,l]
            }
    }
}

// ---------------- sliding-window attention ----------------
// grid (L/64=32, B*H=32), 128 threads = 4 waves; each wave owns one 16-query block.
__global__ void __launch_bounds__(128) attn_kernel(
    const _Float16* __restrict__ qh, const _Float16* __restrict__ kh,
    const _Float16* __restrict__ vT, _Float16* __restrict__ attn) {
    __shared__ _Float16 Pl[4][16][160];  // per-wave P tile, 144 real cols + 16 zero pad

    const int lane   = threadIdx.x & 31;
    const int wave   = threadIdx.x >> 5;
    const int lane15 = lane & 15;
    const int halfw  = lane >> 4;
    const int base   = halfw * 8;
    const int bh     = blockIdx.y;
    const int q0     = blockIdx.x * 64 + wave * 16;

    const _Float16* Qp = qh + (size_t)bh * LSEQ * DH;
    const _Float16* Kp = kh + (size_t)bh * LSEQ * DH;
    const _Float16* Vt = vT + (size_t)bh * DH * LSEQ;

    const _Float16* qrow = Qp + (size_t)(q0 + lane15) * DH;
    v16h qf0 = load_frag(qrow, base);        // d = 0..31
    v16h qf1 = load_frag(qrow + 32, base);   // d = 32..63

    // ---- scores S = (Q/sqrt(Dh)) K^T over 9 key-chunks covering [q0-64, q0+80) ----
    v8f S[9];
#pragma unroll
    for (int c = 0; c < 9; ++c) {
        int j0   = q0 - WIN + c * 16;
        int kidx = j0 + lane15;                              // this lane's key column
        int kcl  = kidx < 0 ? 0 : (kidx > LSEQ - 1 ? LSEQ - 1 : kidx);
        const _Float16* krow = Kp + (size_t)kcl * DH;
        v8f z = {};
        v8f acc = wmma_f16(qf0, load_frag(krow, base), z);
        acc = wmma_f16(qf1, load_frag(krow + 32, base), acc);
#pragma unroll
        for (int r = 0; r < 8; ++r) {                        // mask in accumulator layout
            int qi = q0 + r + halfw * 8;
            int dd = qi - kidx; if (dd < 0) dd = -dd;
            bool ok = (kidx >= 0) && (kidx < LSEQ) && (dd <= WIN);
            acc[r] = ok ? acc[r] : -__builtin_inff();
        }
        S[c] = acc;
    }

    // ---- softmax across the 144-wide window (rows live in 16-lane halves) ----
    float m8[8], s8[8];
#pragma unroll
    for (int r = 0; r < 8; ++r) m8[r] = -__builtin_inff();
#pragma unroll
    for (int c = 0; c < 9; ++c)
#pragma unroll
        for (int r = 0; r < 8; ++r) m8[r] = fmaxf(m8[r], S[c][r]);
#pragma unroll
    for (int r = 0; r < 8; ++r) {
        float m = m8[r];
        m = fmaxf(m, __shfl_xor(m, 1, 32));
        m = fmaxf(m, __shfl_xor(m, 2, 32));
        m = fmaxf(m, __shfl_xor(m, 4, 32));
        m = fmaxf(m, __shfl_xor(m, 8, 32));   // xor masks <16 stay within each half
        m8[r] = m;
        s8[r] = 0.f;
    }
#pragma unroll
    for (int c = 0; c < 9; ++c)
#pragma unroll
        for (int r = 0; r < 8; ++r) {
            float pv = __expf(S[c][r] - m8[r]);  // masked -> exp(-inf) = 0
            S[c][r] = pv;
            s8[r] += pv;
        }
#pragma unroll
    for (int r = 0; r < 8; ++r) {
        float s = s8[r];
        s += __shfl_xor(s, 1, 32);
        s += __shfl_xor(s, 2, 32);
        s += __shfl_xor(s, 4, 32);
        s += __shfl_xor(s, 8, 32);
        s8[r] = 1.0f / s;
    }

    // ---- P (C-layout) -> LDS, re-read as A-fragments ----
#pragma unroll
    for (int c = 0; c < 9; ++c)
#pragma unroll
        for (int r = 0; r < 8; ++r)
            Pl[wave][r + halfw * 8][c * 16 + lane15] = (_Float16)S[c][r];
    { v8h z = {}; *(v8h*)&Pl[wave][lane15][144 + base] = z; }  // zero pad cols 144..159
    __syncthreads();

    // ---- O = P * V (K = 160 padded, 5 WMMA k-steps, 4 N-tiles of 16 dims) ----
    v8f o[4];
#pragma unroll
    for (int kk = 0; kk < 5; ++kk) {
        const _Float16* prow = &Pl[wave][lane15][0];
        v16h pa = load_frag(prow + kk * 32, base);           // ds_load_b128 x2
        int kb = q0 - WIN + kk * 32 + base;
        int s0 = kb;      if (s0 < 0) s0 = 0; if (s0 > LSEQ - 8) s0 = LSEQ - 8;
        int s1 = kb + 16; if (s1 < 0) s1 = 0; if (s1 > LSEQ - 8) s1 = LSEQ - 8;
#pragma unroll
        for (int t = 0; t < 4; ++t) {
            const _Float16* vrow = Vt + (size_t)(t * 16 + lane15) * LSEQ;
            v16h vb = make_frag(vrow + s0, vrow + s1);       // contiguous: V stored transposed
            if (kk == 0) { v8f z = {}; o[t] = wmma_f16(pa, vb, z); }
            else         { o[t] = wmma_f16(pa, vb, o[t]); }
        }
    }

    // ---- normalize rows and emit [B, L, H*Dh] f16 for the output GEMM ----
    int b = bh >> 3, h = bh & 7;
#pragma unroll
    for (int t = 0; t < 4; ++t)
#pragma unroll
        for (int r = 0; r < 8; ++r) {
            int l = q0 + r + halfw * 8;
            int d = t * 16 + lane15;
            attn[((size_t)b * LSEQ + l) * DM + h * DH + d] = (_Float16)(o[t][r] * s8[r]);
        }
}

// ---------------- output projection GEMM (fp32 out) ----------------
// grid (8, 64), 128 threads = 4 waves; 32x64 per wave, ping-pong pipelined.
__global__ void __launch_bounds__(128) o_gemm_kernel(
    const _Float16* __restrict__ ah, const _Float16* __restrict__ woT,
    const float* __restrict__ bo, float* __restrict__ out) {
    const int lane   = threadIdx.x & 31;
    const int wave   = threadIdx.x >> 5;
    const int lane15 = lane & 15;
    const int halfw  = lane >> 4;
    const int base   = halfw * 8;
    const int n0     = blockIdx.x * 64;
    const int m0     = blockIdx.y * 128 + wave * 32;

    const _Float16* arow0 = ah + (size_t)(m0 + lane15) * DM;
    const _Float16* arow1 = arow0 + (size_t)16 * DM;
    const _Float16* brow[4];
#pragma unroll
    for (int t = 0; t < 4; ++t)
        brow[t] = woT + (size_t)(n0 + t * 16 + lane15) * DM;

    v8f acc[2][4];
    v16h a0[2], b0[4], a1[2], b1[4];
    load_ab(a0, b0, arow0, arow1, brow, 0, base);
    load_ab(a1, b1, arow0, arow1, brow, 32, base);
    mma_first(acc, a0, b0);

    for (int k0 = 64; k0 < DM; k0 += 64) {
        load_ab(a0, b0, arow0, arow1, brow, k0, base);
        mma_step(acc, a1, b1);
        load_ab(a1, b1, arow0, arow1, brow, k0 + 32, base);
        mma_step(acc, a0, b0);
    }
    mma_step(acc, a1, b1);

#pragma unroll
    for (int t = 0; t < 4; ++t) {
        int col = n0 + t * 16 + lane15;
        float bcol = bo[col];
#pragma unroll
        for (int rb = 0; rb < 2; ++rb)
#pragma unroll
            for (int r = 0; r < 8; ++r) {
                int m = m0 + rb * 16 + r + halfw * 8;
                out[(size_t)m * DM + col] = acc[rb][t][r] + bcol;
            }
    }
}

extern "C" void kernel_launch(void* const* d_in, const int* in_sizes, int n_in,
                              void* d_out, int out_size, void* d_ws, size_t ws_size,
                              hipStream_t stream) {
    (void)in_sizes; (void)n_in; (void)out_size; (void)ws_size;
    const float* x  = (const float*)d_in[0];
    const float* Wq = (const float*)d_in[1];
    const float* bq = (const float*)d_in[2];
    const float* Wk = (const float*)d_in[3];
    const float* bk = (const float*)d_in[4];
    const float* Wv = (const float*)d_in[5];
    const float* bv = (const float*)d_in[6];
    const float* Wo = (const float*)d_in[7];
    const float* bo = (const float*)d_in[8];
    float* out = (float*)d_out;

    char* ws = (char*)d_ws;
    size_t off = 0;
    auto alloc = [&](size_t bytes) { void* p = ws + off; off += (bytes + 255) & ~(size_t)255; return p; };
    _Float16* xh    = (_Float16*)alloc((size_t)MTOT * DM * 2);
    _Float16* wqkvT = (_Float16*)alloc((size_t)3 * DM * DM * 2);
    _Float16* woT   = (_Float16*)alloc((size_t)DM * DM * 2);
    _Float16* qh    = (_Float16*)alloc((size_t)MTOT * DM * 2);
    _Float16* kh    = (_Float16*)alloc((size_t)MTOT * DM * 2);
    _Float16* vT    = (_Float16*)alloc((size_t)MTOT * DM * 2);
    _Float16* ah    = (_Float16*)alloc((size_t)MTOT * DM * 2);

    cvt_x_kernel<<<2048, 256, 0, stream>>>(x, xh, MTOT * DM);
    cvt_w_kernel<<<4096, 256, 0, stream>>>(Wq, Wk, Wv, Wo, wqkvT, woT);
    qkv_gemm_kernel<<<dim3(8, 64, 3), 128, 0, stream>>>(xh, wqkvT, bq, bk, bv, qh, kh, vT);
    attn_kernel<<<dim3(32, 32), 128, 0, stream>>>(qh, kh, vT, ah);
    o_gemm_kernel<<<dim3(8, 64), 128, 0, stream>>>(ah, woT, bo, out);
}
